// Decompressor_32581621907863
// MI455X (gfx1250) — matrix operand order repaired and checked by reading, not compiled, not run on hardware
//
#include <hip/hip_runtime.h>
#include <hip/hip_bf16.h>
#include <math.h>

typedef __bf16 bf16_t;
typedef __attribute__((ext_vector_type(16))) __bf16 v16bf;
typedef __attribute__((ext_vector_type(8)))  float  v8f;

constexpr int NB = 128;    // batch
constexpr int NP = 1024;   // hidden / parent embed
constexpr int NE = 1024;   // output embed
constexpr int NT = 256;    // sequence length
constexpr int G4 = 4 * NP; // 4096 gate columns
constexpr int KC = 32;     // K-chunk (one WMMA K)
constexpr int LROW = 24;   // padded LDS row stride in elements (48 B)
constexpr int LBUF = 4 * KC * LROW;  // elements per LDS buffer (6 KB)

union Frag {
    v16bf v;
    uint4 q[2];
};

__device__ __forceinline__ float sigmoid_(float x) {
    return 1.0f / (1.0f + __expf(-x));
}

// --- CDNA5 async global->LDS copy (ASYNCcnt-tracked), 16 B per lane --------
// VDST operand carries the wave-relative LDS byte offset; for a __shared__
// object the low 32 bits of its generic (flat) address are exactly that
// offset.
__device__ __forceinline__ void async_cp_b128(const void* gsrc, void* ldst) {
    unsigned loff = (unsigned)(unsigned long long)(ldst);
    unsigned long long gaddr = (unsigned long long)(gsrc);
    asm volatile("global_load_async_to_lds_b128 %0, %1, off"
                 :: "v"(loff), "v"(gaddr) : "memory");
}

__device__ __forceinline__ void wait_async0() {
#if __has_builtin(__builtin_amdgcn_s_wait_asynccnt)
    __builtin_amdgcn_s_wait_asynccnt(0);
#else
    asm volatile("s_wait_asynccnt 0x0" ::: "memory");
#endif
}

// ---------------------------------------------------------------------------
// Prep kernels: transpose weights into K-major bf16, fold biases, init state.
// ---------------------------------------------------------------------------
__global__ __launch_bounds__(256) void prep_gate_weights(
    const float* __restrict__ Wih, const float* __restrict__ Whh,
    bf16_t* __restrict__ WihT, bf16_t* __restrict__ WcT) {
    int o = blockIdx.x * blockDim.x + threadIdx.x;  // over 4096*1024
    int k = o >> 12;
    int j = o & 4095;
    float wi = Wih[j * NP + k];
    float wh = Whh[j * NP + k];
    WihT[o] = (bf16_t)wi;
    WcT[o]  = (bf16_t)(wi + wh);
}

__global__ __launch_bounds__(256) void prep_out_weights(
    const float* __restrict__ Wout, bf16_t* __restrict__ WoutT) {
    int o = blockIdx.x * blockDim.x + threadIdx.x;  // over 1024*1024
    int k = o >> 10;
    int e = o & 1023;
    WoutT[o] = (bf16_t)Wout[e * NP + k];
}

__global__ __launch_bounds__(256) void prep_misc(
    const float* __restrict__ x, const float* __restrict__ bih,
    const float* __restrict__ bhh, float* __restrict__ bias,
    bf16_t* __restrict__ xbf, float* __restrict__ c) {
    int o = blockIdx.x * blockDim.x + threadIdx.x;  // over 131072
    if (o < G4) bias[o] = bih[o] + bhh[o];
    xbf[o] = (bf16_t)x[o];
    c[o] = 0.0f;
}

// ---------------------------------------------------------------------------
// One recurrent step, fused: gates GEMM (WMMA bf16, async-LDS staged B)
// + LSTM cell.  grid = 64 blocks (16 p-cols each), 256 threads = 8 waves.
// Per K-chunk the block stages 4 KB of W^T once (256 lanes x b128 async),
// double-buffered; all 8 waves consume it from LDS.
// ---------------------------------------------------------------------------
__global__ __launch_bounds__(256) void lstm_step(
    const bf16_t* __restrict__ xin,   // [NB, NP] bf16
    const bf16_t* __restrict__ WT,    // [NP, 4NP] bf16, K-major
    const float*  __restrict__ bias,  // [4NP]
    float*        __restrict__ c,     // [NB, NP] f32 state
    bf16_t*       __restrict__ hout)  // [NB, NP] bf16
{
    __shared__ __align__(16) bf16_t sB[2 * LBUF];  // 2 x 6 KB

    const int lane = threadIdx.x & 31;
    const int wave = threadIdx.x >> 5;
    const int half = lane >> 4;
    const int lm   = lane & 15;
    const int rowBase = wave * 16;   // 8 waves cover the 128 batch rows
    const int p0 = blockIdx.x * 16;  // 64 blocks cover 1024 p-cols

    // Per-thread staging slot: one b128 of the 4 KB chunk.
    const int sn   = threadIdx.x >> 6;        // gate tile 0..3
    const int ssub = threadIdx.x & 63;
    const int sk   = ssub >> 1;               // K row 0..31
    const int shi  = (ssub & 1) * 8;          // col half 0 / 8
    const bf16_t* sgbase = WT + (size_t)sk * G4 + sn * NP + p0 + shi;
    bf16_t* sdst = &sB[(sn * KC + sk) * LROW + shi];

    v8f acc[4];
    v8f zero = {0.f, 0.f, 0.f, 0.f, 0.f, 0.f, 0.f, 0.f};
#pragma unroll
    for (int g = 0; g < 4; ++g) acc[g] = zero;

    const bf16_t* aptr = xin + (size_t)(rowBase + lm) * NP;

    async_cp_b128(sgbase, sdst);  // stage chunk 0 into buffer 0

    int buf = 0;
#pragma unroll 1
    for (int kt = 0; kt < NP / KC; ++kt) {
        const int k0 = kt * KC;
        const int kn = (k0 + KC) & (NP - 1);  // wraps to 0 on last iteration

        // A fragment (16x32 bf16) direct from global; warm next chunk.
        Frag a;
        a.q[0] = *(const uint4*)(aptr + k0 + half * 8);
        a.q[1] = *(const uint4*)(aptr + k0 + 16 + half * 8);
        __builtin_prefetch(aptr + kn, 0, 3);

        wait_async0();     // our slice of current buf is in LDS
        __syncthreads();   // everyone's slice landed; prev-buf reads done
        // Unconditionally stage the next chunk (last iter: harmless refetch
        // of chunk 0 into the dead buffer; S_ENDPGM waits all counters).
        async_cp_b128(sgbase + (size_t)kn * G4, sdst + (buf ^ 1) * LBUF);

        const bf16_t* bbase = sB + buf * LBUF + lm;  // B col N = lm
        Frag b0, b1, b2, b3;
#pragma unroll
        for (int s = 0; s < 16; ++s) {
            const int kk = (s < 8) ? (half * 8 + s) : (16 + half * 8 + (s - 8));
            b0.v[s] = bbase[0 * (KC * LROW) + kk * LROW];
            b1.v[s] = bbase[1 * (KC * LROW) + kk * LROW];
            b2.v[s] = bbase[2 * (KC * LROW) + kk * LROW];
            b3.v[s] = bbase[3 * (KC * LROW) + kk * LROW];
        }

        acc[0] = __builtin_amdgcn_wmma_f32_16x16x32_bf16(
            false, a.v, false, b0.v, (short)0, acc[0], false, false);
        acc[1] = __builtin_amdgcn_wmma_f32_16x16x32_bf16(
            false, a.v, false, b1.v, (short)0, acc[1], false, false);
        acc[2] = __builtin_amdgcn_wmma_f32_16x16x32_bf16(
            false, a.v, false, b2.v, (short)0, acc[2], false, false);
        acc[3] = __builtin_amdgcn_wmma_f32_16x16x32_bf16(
            false, a.v, false, b3.v, (short)0, acc[3], false, false);
        buf ^= 1;
    }

    // Epilogue: LSTM cell on this block's [128 x 16] p-slice.
    const int col = p0 + lm;
    const float bi = bias[0 * NP + col];
    const float bf = bias[1 * NP + col];
    const float bg = bias[2 * NP + col];
    const float bo = bias[3 * NP + col];

#pragma unroll
    for (int r = 0; r < 8; ++r) {
        const int m = rowBase + half * 8 + r;  // C/D VGPR r -> row per ISA
        const float iv = sigmoid_(acc[0][r] + bi);
        const float fv = sigmoid_(acc[1][r] + bf);
        const float gv = tanhf(acc[2][r] + bg);
        const float ov = sigmoid_(acc[3][r] + bo);
        const float cold = c[m * NP + col];
        const float cn = fv * cold + iv * gv;
        c[m * NP + col] = cn;
        hout[m * NP + col] = (bf16_t)(ov * tanhf(cn));
    }
}

// ---------------------------------------------------------------------------
// Output projection: [T*B, P] x [P, E] -> transposed [B, T, E] + bias.
// grid = (E/64, 32768/128); same async-LDS staged B pipeline (4 KB chunks).
// ---------------------------------------------------------------------------
__global__ __launch_bounds__(256) void out_gemm(
    const bf16_t* __restrict__ ys,     // [NT*NB, NP] bf16 (t-major rows)
    const bf16_t* __restrict__ WoutT,  // [NP, NE] bf16, K-major
    const float*  __restrict__ bout,   // [NE]
    float*        __restrict__ out)    // [NB, NT, NE] f32
{
    __shared__ __align__(16) bf16_t sB[2 * LBUF];

    const int lane = threadIdx.x & 31;
    const int wave = threadIdx.x >> 5;
    const int half = lane >> 4;
    const int lm   = lane & 15;

    const int rowBase = blockIdx.y * 128 + wave * 16;  // over 32768 rows
    const int colBase = blockIdx.x * 64;               // over 1024 cols

    const int sn   = threadIdx.x >> 6;        // col tile 0..3
    const int ssub = threadIdx.x & 63;
    const int sk   = ssub >> 1;
    const int shi  = (ssub & 1) * 8;
    const bf16_t* sgbase = WoutT + (size_t)sk * NE + colBase + sn * 16 + shi;
    bf16_t* sdst = &sB[(sn * KC + sk) * LROW + shi];

    v8f acc[4];
    v8f zero = {0.f, 0.f, 0.f, 0.f, 0.f, 0.f, 0.f, 0.f};
#pragma unroll
    for (int n = 0; n < 4; ++n) acc[n] = zero;

    const bf16_t* aptr = ys + (size_t)(rowBase + lm) * NP;

    async_cp_b128(sgbase, sdst);

    int buf = 0;
#pragma unroll 1
    for (int kt = 0; kt < NP / KC; ++kt) {
        const int k0 = kt * KC;
        const int kn = (k0 + KC) & (NP - 1);

        Frag a;
        a.q[0] = *(const uint4*)(aptr + k0 + half * 8);
        a.q[1] = *(const uint4*)(aptr + k0 + 16 + half * 8);
        __builtin_prefetch(aptr + kn, 0, 3);

        wait_async0();
        __syncthreads();
        async_cp_b128(sgbase + (size_t)kn * NE, sdst + (buf ^ 1) * LBUF);

        const bf16_t* bbase = sB + buf * LBUF + lm;
        Frag b0, b1, b2, b3;
#pragma unroll
        for (int s = 0; s < 16; ++s) {
            const int kk = (s < 8) ? (half * 8 + s) : (16 + half * 8 + (s - 8));
            b0.v[s] = bbase[0 * (KC * LROW) + kk * LROW];
            b1.v[s] = bbase[1 * (KC * LROW) + kk * LROW];
            b2.v[s] = bbase[2 * (KC * LROW) + kk * LROW];
            b3.v[s] = bbase[3 * (KC * LROW) + kk * LROW];
        }

        acc[0] = __builtin_amdgcn_wmma_f32_16x16x32_bf16(
            false, a.v, false, b0.v, (short)0, acc[0], false, false);
        acc[1] = __builtin_amdgcn_wmma_f32_16x16x32_bf16(
            false, a.v, false, b1.v, (short)0, acc[1], false, false);
        acc[2] = __builtin_amdgcn_wmma_f32_16x16x32_bf16(
            false, a.v, false, b2.v, (short)0, acc[2], false, false);
        acc[3] = __builtin_amdgcn_wmma_f32_16x16x32_bf16(
            false, a.v, false, b3.v, (short)0, acc[3], false, false);
        buf ^= 1;
    }

#pragma unroll
    for (int n = 0; n < 4; ++n) {
        const int col = colBase + n * 16 + lm;
        const float bb = bout[col];
#pragma unroll
        for (int r = 0; r < 8; ++r) {
            const int row = rowBase + half * 8 + r;  // row = t*128 + b
            const int t = row >> 7;
            const int b = row & 127;
            out[((size_t)b * NT + t) * NE + col] = acc[n][r] + bb;
        }
    }
}

// ---------------------------------------------------------------------------
extern "C" void kernel_launch(void* const* d_in, const int* in_sizes, int n_in,
                              void* d_out, int out_size, void* d_ws, size_t ws_size,
                              hipStream_t stream) {
    (void)in_sizes; (void)n_in; (void)out_size; (void)ws_size;

    const float* x    = (const float*)d_in[0];
    const float* Wih  = (const float*)d_in[1];
    const float* Whh  = (const float*)d_in[2];
    const float* bih  = (const float*)d_in[3];
    const float* bhh  = (const float*)d_in[4];
    const float* Wout = (const float*)d_in[5];
    const float* bout = (const float*)d_in[6];
    float* out = (float*)d_out;

    // Workspace layout (all offsets 256-aligned); total ~86.8 MB.
    char* ws = (char*)d_ws;
    bf16_t* WcT   = (bf16_t*)(ws);             //  8 MB: (Wih+Whh)^T bf16
    bf16_t* WihT  = (bf16_t*)(ws + 8388608);   //  8 MB: Wih^T bf16
    bf16_t* WoutT = (bf16_t*)(ws + 16777216);  //  2 MB: Wout^T bf16
    float*  bias  = (float*) (ws + 18874368);  // 16 KB: bih+bhh
    float*  cst   = (float*) (ws + 18890752);  // 512 KB: cell state f32
    bf16_t* xbf   = (bf16_t*)(ws + 19415040);  // 256 KB: x bf16
    bf16_t* ysb   = (bf16_t*)(ws + 19677184);  // 64 MB: ys [T,B,P] bf16

    prep_gate_weights<<<(G4 * NP) / 256, 256, 0, stream>>>(Wih, Whh, WihT, WcT);
    prep_out_weights<<<(NP * NE) / 256, 256, 0, stream>>>(Wout, WoutT);
    prep_misc<<<(NB * NP) / 256, 256, 0, stream>>>(x, bih, bhh, bias, xbf, cst);

    // Sequential recurrence: step 0 uses x and W_ih (h0 == 0); steps >= 1
    // exploit xin == h so one GEMM against the combined weight suffices.
    for (int t = 0; t < NT; ++t) {
        const bf16_t* xin = (t == 0) ? xbf : (ysb + (size_t)(t - 1) * NB * NP);
        const bf16_t* WT  = (t == 0) ? WihT : WcT;
        lstm_step<<<NP / 16, 256, 0, stream>>>(xin, WT, bias, cst,
                                               ysb + (size_t)t * NB * NP);
    }

    out_gemm<<<dim3(NE / 64, (NT * NB) / 128), 256, 0, stream>>>(
        ysb, WoutT, bout, out);
}
